// EdgeAttr_21620865368394
// MI455X (gfx1250) — compile-verified, matrix operand order checked by reading, not run on hardware
//
#include <hip/hip_runtime.h>
#include <math.h>

// CDNA5 wave32 WMMA types
typedef __attribute__((ext_vector_type(16))) _Float16 v16h;
typedef __attribute__((ext_vector_type(8)))  _Float16 v8h;
typedef __attribute__((ext_vector_type(8)))  float    v8f;

#define D_COUNT 16
#define HIDDEN  128
#define TILE_M  16          // edges per wave-tile
#define WAVES_PER_BLOCK 8   // 256 threads
#define SROW    132         // LDS staging row stride in floats (128 + 4 pad, conflict-free)
#define KDIM    32          // K padded for 16x16x32 WMMA: 0..15 = W, 16 = bias, 17..31 = 0
#define SWS     40          // sWt column stride in halves (80 B: 16B-aligned, conflict-free)

// RBF constants: mu = linspace(0,6,16) -> mu_k = 0.4*k ; sigma = 6/16 = 0.375
#define MU_STEP   0.4f
#define INV_SIGMA (1.0f / 0.375f)

// sigmoid: prefer single-trans v_tanh_f32 if the toolchain exposes it,
// else native v_exp_f32 + v_rcp_f32 (2 trans ops).
__device__ __forceinline__ float fast_sigmoid(float x) {
#if __has_builtin(__builtin_amdgcn_tanhf)
    return fmaf(0.5f, __builtin_amdgcn_tanhf(0.5f * x), 0.5f);
#else
    return __builtin_amdgcn_rcpf(1.0f + __expf(-x));
#endif
}

__global__ __launch_bounds__(256)
void edge_attr_wmma_kernel(const float* __restrict__ pos,
                           const long long* __restrict__ edge_index,
                           const float* __restrict__ W,
                           const float* __restrict__ bias,
                           float* __restrict__ out,
                           int n_edges)
{
    // Transposed weights in LDS: per output column, contiguous K-vector of 32 halves:
    //   k = 0..15 : W[k][col]   (real weights)
    //   k = 16    : bias[col]   (bias folded into the GEMM via A's K=16 lane = 1.0)
    //   k = 17..31: 0           (padding)
    __shared__ __align__(16) _Float16 sWt[HIDDEN * SWS];                   // 10 KB
    __shared__ __align__(16) float sOut[WAVES_PER_BLOCK][TILE_M * SROW];   // 66 KB

    if (threadIdx.x < HIDDEN) {
        const int col = threadIdx.x;
        _Float16* c = &sWt[col * SWS];
#pragma unroll
        for (int k = 0; k < D_COUNT; ++k) c[k] = (_Float16)W[k * HIDDEN + col];
        c[16] = (_Float16)bias[col];
#pragma unroll
        for (int k = 17; k < KDIM; ++k) c[k] = (_Float16)0.0f;
    }
    __syncthreads();

    const int lane = threadIdx.x & 31;
    const int wave = threadIdx.x >> 5;
    const int m    = lane & 15;   // row within 16-edge tile (also B/C column index)
    const int half = lane >> 4;   // 0: lanes 0-15, 1: lanes 16-31

    const long long tile = (long long)blockIdx.x * WAVES_PER_BLOCK + wave;
    const long long e0   = tile * TILE_M;
    if (e0 >= (long long)n_edges) return;   // wave-uniform exit (after syncthreads)

    // ---- per-lane edge distance (lanes L and L+16 duplicate edge m) ----
    const long long e  = e0 + m;
    const long long ec = (e < (long long)n_edges) ? e : (long long)(n_edges - 1); // select, not branch

    const long long src = edge_index[ec];                       // edge_index[0][e]
    const long long dst = edge_index[(long long)n_edges + ec];  // edge_index[1][e]
    const float dx = pos[src * 3 + 0] - pos[dst * 3 + 0];
    const float dy = pos[src * 3 + 1] - pos[dst * 3 + 1];
    const float dz = pos[src * 3 + 2] - pos[dst * 3 + 2];
    const float dist = sqrtf(dx * dx + dy * dy + dz * dz);

    // ---- A matrix: 16x32 f16. element i (0..7) = rbf(K = half*8 + i);
    //      element 8 = 1.0 on half 0 (K=16, the bias lane); rest zero. ----
    v16h a;
#pragma unroll
    for (int i = 0; i < 8; ++i) {
        const float tt = (dist - MU_STEP * (float)(half * 8 + i)) * INV_SIGMA;
        a[i] = (_Float16)__expf(-tt * tt);
    }
    a[8] = (half == 0) ? (_Float16)1.0f : (_Float16)0.0f;
#pragma unroll
    for (int i = 9; i < 16; ++i) a[i] = (_Float16)0.0f;

    // ---- 8 N-tiles: D = A x [W;bias;0]_tile, then sigmoid -> LDS staging ----
    const int n = m;                                   // lane's column within an N-tile
    const _Float16* wb = &sWt[n * SWS + half * 8];     // per-lane B fragment base
    float* srow = &sOut[wave][(8 * half) * SROW + n];  // per-lane staging base

#pragma unroll
    for (int t = 0; t < 8; ++t) {
        // B fragment: elements 0..7 = K(half*8 .. +7), 8..15 = K(16+half*8 .. +7)
        const v8h lo = *(const v8h*)(wb + t * 16 * SWS);        // ds_load_b128
        const v8h hi = *(const v8h*)(wb + t * 16 * SWS + 16);   // ds_load_b128
        v16h bm;
#pragma unroll
        for (int i = 0; i < 8; ++i) { bm[i] = lo[i]; bm[i + 8] = hi[i]; }

        v8f c = {};
        c = __builtin_amdgcn_wmma_f32_16x16x32_f16(
                /*neg_a=*/false, a, /*neg_b=*/false, bm,
                /*c_mod=*/(short)0, c, /*reuse_a=*/false, /*reuse_b=*/false);

        // C/D layout: VGPR r holds row m = r + 8*half; bias already inside c[r].
#pragma unroll
        for (int r = 0; r < 8; ++r) {
            const float s = fast_sigmoid(c[r]);
            srow[r * SROW + t * 16] = s;                        // ds_store_b32, imm offset
        }
    }

    // ---- coalesced writeback: 16 rows x 128 floats, b128 per lane per row ----
    // Same-wave LDS ops are ordered; compiler inserts the dscnt wait.
    float*       outp  = out + e0 * (long long)HIDDEN + lane * 4;
    const float* sload = &sOut[wave][lane * 4];
    const bool full = (e0 + TILE_M) <= (long long)n_edges;      // wave-uniform
    if (full) {
#pragma unroll
        for (int i = 0; i < TILE_M; ++i)
            *(float4*)(outp + i * HIDDEN) = *(const float4*)(sload + i * SROW);
    } else {
        for (int i = 0; i < TILE_M; ++i) {
            if (e0 + i < (long long)n_edges)                    // uniform per row
                *(float4*)(outp + i * HIDDEN) = *(const float4*)(sload + i * SROW);
        }
    }
}

extern "C" void kernel_launch(void* const* d_in, const int* in_sizes, int n_in,
                              void* d_out, int out_size, void* d_ws, size_t ws_size,
                              hipStream_t stream) {
    (void)n_in; (void)d_ws; (void)ws_size; (void)out_size;
    const float*     pos = (const float*)d_in[0];
    const long long* ei  = (const long long*)d_in[1];   // int64 [2, E]
    const float*     W   = (const float*)d_in[2];       // [16, 128]
    const float*     b   = (const float*)d_in[3];       // [128]
    float*           out = (float*)d_out;               // [E, 128]

    const int n_edges = in_sizes[1] / 2;                // 1,600,000
    const long long tiles  = ((long long)n_edges + TILE_M - 1) / TILE_M;
    const int       blocks = (int)((tiles + WAVES_PER_BLOCK - 1) / WAVES_PER_BLOCK);

    edge_attr_wmma_kernel<<<blocks, 256, 0, stream>>>(pos, ei, W, b, out, n_edges);
}
